// PointCloudPerceiverChannelsEncoder_49271864819782
// MI455X (gfx1250) — matrix operand order repaired and checked by reading, not compiled, not run on hardware
//
#include <hip/hip_runtime.h>
#include <hip/hip_bf16.h>

typedef _Float16 h16;
typedef _Float16 v16h __attribute__((ext_vector_type(16)));
typedef float    v8f  __attribute__((ext_vector_type(8)));

#define DI __device__ __forceinline__

#ifndef USE_ASYNC_LDS
#define USE_ASYNC_LDS 1     // gfx1250 global_load_async_to_lds_b128 path
#endif

// ---------------------------------------------------------------------------
// WMMA fragment helpers (gfx1250 wave32, v_wmma_f32_16x16x32_f16)
// A 16x32 f16: lane holds row m = lane%16; halves [k0+8h .. k0+8h+7] and
// [k0+16+8h .. k0+16+8h+7]  (h = lane>>4)  -> two contiguous 16B loads.
// B 32x16 f16: lane holds col n = lane%16; halves [k0+8h .. k0+8h+7] and
// [k0+16+8h .. k0+16+8h+7]  (h = lane>>4)  -> two contiguous 16B loads.
// C/D f32: lane holds col n = lane%16, rows j + 8*(lane>>4), j = 0..7.
// ---------------------------------------------------------------------------
DI v16h load_a_frag(const h16* A, int ld, int m0, int k0, int lane) {
  int m = m0 + (lane & 15);
  int h = lane >> 4;
  const h16* p = A + (size_t)m * ld + k0 + 8 * h;
  union { v16h v; uint4 u[2]; } f;
  f.u[0] = *(const uint4*)(p);
  f.u[1] = *(const uint4*)(p + 16);
  return f.v;
}
DI v16h load_b_frag(const h16* Bt, int ld, int n0, int k0, int lane) {
  int n = n0 + (lane & 15);
  const h16* p = Bt + (size_t)n * ld + k0 + 8 * (lane >> 4);
  union { v16h v; uint4 u[2]; } f;
  f.u[0] = *(const uint4*)(p);
  f.u[1] = *(const uint4*)(p + 16);
  return f.v;
}
DI v8f wmma16(v16h a, v16h b, v8f c) {
  return __builtin_amdgcn_wmma_f32_16x16x32_f16(false, a, false, b, (short)0, c,
                                                false, false);
}

// ---------------------------------------------------------------------------
// Weight prep: f32 -> fp16, transposed to [N][K] (B-operand layout), or
// cast+zero-pad for the conv weights (already [co][K] in memory).
// ---------------------------------------------------------------------------
__global__ void k_cast_pad(const float* __restrict__ src, h16* __restrict__ dst,
                           int ksrc, int kdst, int rows) {
  int i = blockIdx.x * blockDim.x + threadIdx.x;
  int total = rows * kdst;
  if (i >= total) return;
  int r = i / kdst, c = i - r * kdst;
  dst[i] = (c < ksrc) ? (h16)src[(size_t)r * ksrc + c] : (h16)0.f;
}
__global__ void k_tcast(const float* __restrict__ src, h16* __restrict__ dst,
                        int K, int N) {
  int i = blockIdx.x * blockDim.x + threadIdx.x;
  int total = K * N;
  if (i >= total) return;
  int k = i / N, n = i - k * N;
  dst[(size_t)n * K + k] = (h16)src[i];
}

// ---------------------------------------------------------------------------
// NeRF posenc + (62x128) projection + fused LN(c_ln2) fp16 output for the KV
// GEMM.  Block = 128 threads, 16 rows/block.  pe_w staged in LDS (31.75 KB).
// ---------------------------------------------------------------------------
__global__ void k_emb(const float* __restrict__ pts, const float* __restrict__ peW,
                      const float* __restrict__ peB, const float* __restrict__ lnS,
                      const float* __restrict__ lnB, float* __restrict__ emb,
                      h16* __restrict__ lnkv) {
  __shared__ float sW[62 * 128];
  __shared__ float sF[16][62];
  int tid = threadIdx.x;
  for (int i = tid; i < 62 * 128; i += 128) sW[i] = peW[i];
  int r0 = blockIdx.x * 16;
  for (int i = tid; i < 16 * 62; i += 128) {
    int r = i / 62, f = i - r * 62;
    size_t gp = (size_t)(r0 + r) * 2;
    float x0 = pts[gp], x1 = pts[gp + 1];
    float val;
    if (f < 2) {
      val = f ? x1 : x0;
    } else if (f < 32) {
      int q = f - 2; float x = (q & 1) ? x1 : x0;
      val = sinf(x * (float)(1 << (q >> 1)));
    } else {
      int q = f - 32; float x = (q & 1) ? x1 : x0;
      val = cosf(x * (float)(1 << (q >> 1)));
    }
    sF[r][f] = val;
  }
  __syncthreads();
  int r = tid >> 3, q = tid & 7;        // 8 lanes per row, 16 cols per lane
  float acc[16];
#pragma unroll
  for (int k = 0; k < 16; k++) acc[k] = peB[q * 16 + k];
  for (int f = 0; f < 62; f++) {
    float fv = sF[r][f];
    const float* wrow = &sW[f * 128 + q * 16];
#pragma unroll
    for (int k = 0; k < 16; k++) acc[k] += fv * wrow[k];
  }
  float s = 0.f, s2 = 0.f;
#pragma unroll
  for (int k = 0; k < 16; k++) { s += acc[k]; s2 += acc[k] * acc[k]; }
  for (int m = 1; m < 8; m <<= 1) {     // reduce over the 8 lanes of this row
    s += __shfl_xor(s, m, 32);
    s2 += __shfl_xor(s2, m, 32);
  }
  float mean = s * (1.f / 128.f);
  float var = s2 * (1.f / 128.f) - mean * mean;
  float rs = rsqrtf(var + 1e-5f);
  size_t base = (size_t)(r0 + r) * 128;
#pragma unroll
  for (int k = 0; k < 16; k++) {
    int c = q * 16 + k;
    float v = acc[k];
    emb[base + c] = v;
    lnkv[base + c] = (h16)((v - mean) * rs * lnS[c] + lnB[c]);
  }
}

// ---------------------------------------------------------------------------
// Radius grouping: first 8 indices (ascending) with dist^2 <= R^2, padded
// with the first hit.  One wave per query; ballot over wave32.
// ---------------------------------------------------------------------------
__global__ void k_bq(const float* __restrict__ pts, int* __restrict__ idx) {
  int lane = threadIdx.x & 31, w = threadIdx.x >> 5;
  int q = blockIdx.x * 4 + w;           // 0..1023
  int b = q >> 7, i = q & 127;
  size_t pb = (size_t)b * 8192;
  float cx = pts[(pb + i) * 2], cy = pts[(pb + i) * 2 + 1];
  int out8[8];
  int cnt = 0;
  for (int j0 = 0; j0 < 8192 && cnt < 8; j0 += 32) {
    int j = j0 + lane;
    float dx = pts[(pb + j) * 2] - cx;
    float dy = pts[(pb + j) * 2 + 1] - cy;
    unsigned mask = (unsigned)__ballot(dx * dx + dy * dy <= 0.04f);
    while (mask && cnt < 8) {
      int bit = __ffs(mask) - 1;
      mask &= mask - 1;
      out8[cnt++] = j0 + bit;
    }
  }
  for (int k = cnt; k < 8; k++) out8[k] = out8[0];
  if (lane < 8) idx[(size_t)q * 8 + lane] = out8[lane];
}

// ---------------------------------------------------------------------------
// im2col for conv1: feats[s][c] (s=8 samples, c=130 ch: 2 xyz-diff + 128 emb)
// A1 row = (b*128+p)*9 + h ; col = ci*8 + kh ; sample = (h+kh+4) % 8.
// K padded 1040 -> 1056.
// ---------------------------------------------------------------------------
__global__ void k_im2col1(const float* __restrict__ pts, const float* __restrict__ emb,
                          const int* __restrict__ idx, h16* __restrict__ A1) {
  __shared__ float sF[8][130];
  __shared__ int sI[8];
  int tid = threadIdx.x;
  int blk = blockIdx.x;                 // b*128 + p
  int b = blk >> 7, p = blk & 127;
  if (tid < 8) sI[tid] = idx[(size_t)blk * 8 + tid];
  __syncthreads();
  for (int t = tid; t < 8 * 130; t += 256) {
    int s = t / 130, c = t - s * 130;
    int j = sI[s];
    float v;
    if (c < 2)
      v = pts[((size_t)b * 8192 + j) * 2 + c] - pts[((size_t)b * 8192 + p) * 2 + c];
    else
      v = emb[((size_t)b * 8192 + j) * 128 + (c - 2)];
    sF[s][c] = v;
  }
  __syncthreads();
  size_t rb = (size_t)blk * 9;
  for (int t = tid; t < 9 * 1056; t += 256) {
    int h = t / 1056, c = t - h * 1056;
    int ci = c >> 3, kh = c & 7;
    float v = (ci < 130) ? sF[(h + kh + 4) & 7][ci] : 0.f;
    A1[(rb + h) * 1056 + c] = (h16)v;
  }
}

// im2col for conv2 (input H=9, circular pad 4, out H=10, sample=(h+kh+5)%9).
// The 9x128 fp16 staging tile is pulled into LDS via the CDNA5 async DMA path
// (GLOBAL_LOAD_ASYNC_TO_LDS_B128, tracked by ASYNCcnt) — no VGPR bounce.
__global__ void k_im2col2(const h16* __restrict__ y1, h16* __restrict__ A2) {
  __shared__ h16 sY[9 * 128];
  int tid = threadIdx.x;
  int blk = blockIdx.x;                 // b*128 + p
  size_t yb = (size_t)blk * 9 * 128;
#if USE_ASYNC_LDS
  if (tid < 144) {                      // 144 lanes x 16B = 2304B tile
    unsigned lds = (unsigned)(unsigned long long)(&sY[tid * 8]);
    const h16* g = y1 + yb + (size_t)tid * 8;
    asm volatile("global_load_async_to_lds_b128 %0, %1, off"
                 :: "v"(lds), "v"(g) : "memory");
  }
  asm volatile("s_wait_asynccnt 0" ::: "memory");
  __syncthreads();
#else
  for (int t = tid; t < 9 * 128; t += 256) sY[t] = y1[yb + t];
  __syncthreads();
#endif
  size_t rb = (size_t)blk * 10;
  for (int t = tid; t < 10 * 1024; t += 256) {
    int h = t / 1024, c = t - h * 1024;
    int ci = c >> 3, kh = c & 7;
    int s = (h + kh + 5) % 9;
    A2[(rb + h) * 1024 + c] = sY[s * 128 + ci];
  }
}

// ---------------------------------------------------------------------------
// data_tokens (mean over 10 conv outputs) ++ latent tokens, then LN(ln_pre)
// -> h f32 [B*256][128].  One wave per row.
// ---------------------------------------------------------------------------
__global__ void k_h0(const h16* __restrict__ y2, const float* __restrict__ otok,
                     const float* __restrict__ sc, const float* __restrict__ bi,
                     float* __restrict__ H) {
  int lane = threadIdx.x & 31, w = threadIdx.x >> 5;
  int row = blockIdx.x * 4 + w;         // 0..2047
  int b = row >> 8, rr = row & 255;
  float v[4];
  if (rr < 128) {
    size_t base = (size_t)((b << 7) | rr) * 10 * 128;
#pragma unroll
    for (int k = 0; k < 4; k++) {
      int c = lane + 32 * k;
      float s = 0.f;
      for (int t = 0; t < 10; t++) s += (float)y2[base + (size_t)t * 128 + c];
      v[k] = s * 0.1f;
    }
  } else {
#pragma unroll
    for (int k = 0; k < 4; k++)
      v[k] = otok[(size_t)(rr - 128) * 128 + lane + 32 * k];
  }
  float s = 0.f, s2 = 0.f;
#pragma unroll
  for (int k = 0; k < 4; k++) { s += v[k]; s2 += v[k] * v[k]; }
  for (int m = 1; m < 32; m <<= 1) {
    s += __shfl_xor(s, m, 32);
    s2 += __shfl_xor(s2, m, 32);
  }
  float mean = s * (1.f / 128.f);
  float var = s2 * (1.f / 128.f) - mean * mean;
  float rs = rsqrtf(var + 1e-5f);
#pragma unroll
  for (int k = 0; k < 4; k++) {
    int c = lane + 32 * k;
    H[(size_t)row * 128 + c] = (v[k] - mean) * rs * sc[c] + bi[c];
  }
}

// ---------------------------------------------------------------------------
// Generic LayerNorm: f32 in -> fp16 out.  mode 1 gathers the last 128 of each
// 256-token block (final LN over h[:, -LATENT_D:]).
// ---------------------------------------------------------------------------
__global__ void k_ln(const float* __restrict__ X, const float* __restrict__ sc,
                     const float* __restrict__ bi, h16* __restrict__ Y,
                     int rows, int mode) {
  int lane = threadIdx.x & 31, w = threadIdx.x >> 5;
  int row = blockIdx.x * 4 + w;
  if (row >= rows) return;
  int src = (mode == 1) ? (((row >> 7) * 256) + 128 + (row & 127)) : row;
  const float* x = X + (size_t)src * 128;
  float v[4];
  float s = 0.f, s2 = 0.f;
#pragma unroll
  for (int k = 0; k < 4; k++) {
    v[k] = x[lane + 32 * k];
    s += v[k]; s2 += v[k] * v[k];
  }
  for (int m = 1; m < 32; m <<= 1) {
    s += __shfl_xor(s, m, 32);
    s2 += __shfl_xor(s2, m, 32);
  }
  float mean = s * (1.f / 128.f);
  float var = s2 * (1.f / 128.f) - mean * mean;
  float rs = rsqrtf(var + 1e-5f);
#pragma unroll
  for (int k = 0; k < 4; k++) {
    int c = lane + 32 * k;
    Y[(size_t)row * 128 + c] = (h16)((v[k] - mean) * rs * sc[c] + bi[c]);
  }
}

// ---------------------------------------------------------------------------
// WMMA GEMM: C[M][N] = A[M][K](fp16) * Bt[N][K](fp16) + bias.
// Each wave computes a 16x64 strip (4 accumulators): the A fragment is loaded
// once per k-tile and reused 4x, and the 4 independent WMMAs cover the
// WMMA->WMMA data hazard without stalls.  4 waves / block.
// ---------------------------------------------------------------------------
enum { GM_F16 = 0, GM_SILU, GM_GELU, GM_QS, GM_ADDF, GM_TANH, GM_KV, GM_QKV };

__global__ void k_gemm(const h16* __restrict__ A, const h16* __restrict__ Bt,
                       const float* __restrict__ bias, int M, int K, int N,
                       int mode, h16* o_h, float* o_f, h16* o_k, h16* o_vt,
                       int Tper, float qscale, int ntiles) {
  int lane = threadIdx.x & 31;
  int tile = blockIdx.x * 4 + (threadIdx.x >> 5);
  if (tile >= ntiles) return;
  int ngrp = N >> 6;                    // 64-wide strips
  int mt = tile / ngrp, ng = tile - mt * ngrp;
  int m0 = mt * 16, n0 = ng * 64;
  v8f acc0 = {}, acc1 = {}, acc2 = {}, acc3 = {};
  for (int kt = 0; kt < K; kt += 32) {
    v16h a = load_a_frag(A, K, m0, kt, lane);
    v16h b0 = load_b_frag(Bt, K, n0, kt, lane);
    v16h b1 = load_b_frag(Bt, K, n0 + 16, kt, lane);
    v16h b2 = load_b_frag(Bt, K, n0 + 32, kt, lane);
    v16h b3 = load_b_frag(Bt, K, n0 + 48, kt, lane);
    acc0 = wmma16(a, b0, acc0);
    acc1 = wmma16(a, b1, acc1);
    acc2 = wmma16(a, b2, acc2);
    acc3 = wmma16(a, b3, acc3);
  }
  int hh = lane >> 4;
#pragma unroll
  for (int sgrp = 0; sgrp < 4; sgrp++) {
    v8f acc = (sgrp == 0) ? acc0 : (sgrp == 1) ? acc1 : (sgrp == 2) ? acc2 : acc3;
    int n = n0 + sgrp * 16 + (lane & 15);
    float bv = bias ? bias[n] : 0.f;
#pragma unroll
    for (int j = 0; j < 8; j++) {
      int m = m0 + j + 8 * hh;
      float v = acc[j] + bv;
      if (mode == GM_F16) {
        o_h[(size_t)m * N + n] = (h16)v;
      } else if (mode == GM_SILU) {
        o_h[(size_t)m * N + n] = (h16)(v / (1.f + __expf(-v)));
      } else if (mode == GM_GELU) {
        o_h[(size_t)m * N + n] = (h16)(0.5f * v * (1.f + erff(v * 0.70710678f)));
      } else if (mode == GM_QS) {
        o_h[(size_t)m * N + n] = (h16)(v * qscale);
      } else if (mode == GM_ADDF) {
        o_f[(size_t)m * N + n] += v;
      } else if (mode == GM_TANH) {
        o_f[(size_t)m * N + n] = tanhf(v);
      } else if (mode == GM_KV) {
        if (n < 128) {
          o_k[(size_t)m * 128 + n] = (h16)v;
        } else {
          int b = m / Tper, t = m - b * Tper;
          o_vt[((size_t)b * 128 + (n - 128)) * Tper + t] = (h16)v;
        }
      } else { // GM_QKV
        if (n < 128) {
          o_h[(size_t)m * 128 + n] = (h16)(v * qscale);
        } else if (n < 256) {
          o_k[(size_t)m * 128 + (n - 128)] = (h16)v;
        } else {
          int b = m / Tper, t = m - b * Tper;
          o_vt[((size_t)b * 128 + (n - 256)) * Tper + t] = (h16)v;
        }
      }
    }
  }
}

// ---------------------------------------------------------------------------
// Flash attention, one wave per (b, head, 16-query tile).  dh = 32.
// Q pre-scaled by 1/sqrt(dh).  K row-major [b][Tk][128]; V transposed
// [b][128][Tk].  Online softmax; P tile bounced through LDS to convert
// D-layout -> A-fragment layout for the P x V WMMA.
// ---------------------------------------------------------------------------
__global__ void k_attn(const h16* __restrict__ Q, const h16* __restrict__ Kb,
                       const h16* __restrict__ Vt, h16* __restrict__ O,
                       int Tq, int Tk) {
  __shared__ h16 plds[16 * 32];
  int lane = threadIdx.x;
  int qtiles = Tq >> 4;
  int qt = blockIdx.x % qtiles;
  int hd = (blockIdx.x / qtiles) & 3;
  int b = blockIdx.x / (qtiles * 4);
  int qrow0 = b * Tq + qt * 16;
  const h16* Kbb = Kb + (size_t)b * Tk * 128 + hd * 32;
  const h16* Vtb = Vt + ((size_t)b * 128 + hd * 32) * Tk;
  v16h aq = load_a_frag(Q, 128, qrow0, hd * 32, lane);
  float mrun[8], lrun[8];
#pragma unroll
  for (int j = 0; j < 8; j++) { mrun[j] = -1e30f; lrun[j] = 0.f; }
  v8f o0 = {}, o1 = {};
  v8f zero = {};
  for (int k0 = 0; k0 < Tk; k0 += 32) {
    v16h bk0 = load_b_frag(Kbb, 128, k0, 0, lane);
    v16h bk1 = load_b_frag(Kbb, 128, k0 + 16, 0, lane);
    v8f s0 = wmma16(aq, bk0, zero);
    v8f s1 = wmma16(aq, bk1, zero);
#pragma unroll
    for (int j = 0; j < 8; j++) {
      float t = fmaxf(s0[j], s1[j]);
      t = fmaxf(t, __shfl_xor(t, 1, 32));
      t = fmaxf(t, __shfl_xor(t, 2, 32));
      t = fmaxf(t, __shfl_xor(t, 4, 32));
      t = fmaxf(t, __shfl_xor(t, 8, 32));
      float mn = fmaxf(mrun[j], t);
      float al = __expf(mrun[j] - mn);
      float p0 = __expf(s0[j] - mn);
      float p1 = __expf(s1[j] - mn);
      float rsum = p0 + p1;
      rsum += __shfl_xor(rsum, 1, 32);
      rsum += __shfl_xor(rsum, 2, 32);
      rsum += __shfl_xor(rsum, 4, 32);
      rsum += __shfl_xor(rsum, 8, 32);
      lrun[j] = lrun[j] * al + rsum;
      o0[j] = o0[j] * al;
      o1[j] = o1[j] * al;
      mrun[j] = mn;
      int r = j + 8 * (lane >> 4);
      plds[r * 32 + (lane & 15)] = (h16)p0;
      plds[r * 32 + 16 + (lane & 15)] = (h16)p1;
    }
    // LDS from the same wave completes in order; fence the compiler and drain
    // DScnt before the cross-lane read-back.
    asm volatile("s_wait_dscnt 0" ::: "memory");
    union { v16h v; uint4 u[2]; } fa;
    const h16* pp = &plds[(lane & 15) * 32 + 8 * (lane >> 4)];
    fa.u[0] = *(const uint4*)pp;
    fa.u[1] = *(const uint4*)(pp + 16);
    asm volatile("" ::: "memory");
    v16h bv0 = load_b_frag(Vtb, Tk, 0, k0, lane);
    v16h bv1 = load_b_frag(Vtb + (size_t)16 * Tk, Tk, 0, k0, lane);
    o0 = wmma16(fa.v, bv0, o0);
    o1 = wmma16(fa.v, bv1, o1);
  }
#pragma unroll
  for (int j = 0; j < 8; j++) {
    int m = qrow0 + j + 8 * (lane >> 4);
    float inv = 1.f / lrun[j];
    O[(size_t)m * 128 + hd * 32 + (lane & 15)] = (h16)(o0[j] * inv);
    O[(size_t)m * 128 + hd * 32 + 16 + (lane & 15)] = (h16)(o1[j] * inv);
  }
}

// ---------------------------------------------------------------------------
static void gemm(hipStream_t s, const h16* A, const h16* Bt, const float* bias,
                 int M, int K, int N, int mode, h16* o_h, float* o_f, h16* o_k,
                 h16* o_vt, int Tper, float qscale) {
  int ntiles = (M >> 4) * (N >> 6);     // 16x64 strips per wave
  int blocks = (ntiles + 3) >> 2;
  k_gemm<<<blocks, 128, 0, s>>>(A, Bt, bias, M, K, N, mode, o_h, o_f, o_k,
                                o_vt, Tper, qscale, ntiles);
}

extern "C" void kernel_launch(void* const* d_in, const int* in_sizes, int n_in,
                              void* d_out, int out_size, void* d_ws,
                              size_t ws_size, hipStream_t stream) {
  (void)in_sizes; (void)n_in; (void)out_size; (void)ws_size;
  const float* points     = (const float*)d_in[0];
  const float* pe_w       = (const float*)d_in[1];
  const float* pe_b       = (const float*)d_in[2];
  const float* conv1_w    = (const float*)d_in[3];
  const float* conv1_b    = (const float*)d_in[4];
  const float* conv2_w    = (const float*)d_in[5];
  const float* conv2_b    = (const float*)d_in[6];
  const float* otok       = (const float*)d_in[7];
  const float* ln_pre_s   = (const float*)d_in[8];
  const float* ln_pre_b   = (const float*)d_in[9];
  const float* c_ln1_s    = (const float*)d_in[10];
  const float* c_ln1_b    = (const float*)d_in[11];
  const float* c_ln2_s    = (const float*)d_in[12];
  const float* c_ln2_b    = (const float*)d_in[13];
  const float* c_q_w      = (const float*)d_in[14];
  const float* c_q_b      = (const float*)d_in[15];
  const float* c_kv_w     = (const float*)d_in[16];
  const float* c_kv_b     = (const float*)d_in[17];
  const float* c_proj_w   = (const float*)d_in[18];
  const float* c_proj_b   = (const float*)d_in[19];
  const float* c_ln3_s    = (const float*)d_in[20];
  const float* c_ln3_b    = (const float*)d_in[21];
  const float* c_mlp1_w   = (const float*)d_in[22];
  const float* c_mlp1_b   = (const float*)d_in[23];
  const float* c_mlp2_w   = (const float*)d_in[24];
  const float* c_mlp2_b   = (const float*)d_in[25];
  const float* s_ln1_s    = (const float*)d_in[26];
  const float* s_ln1_b    = (const float*)d_in[27];
  const float* s_qkv_w    = (const float*)d_in[28];
  const float* s_qkv_b    = (const float*)d_in[29];
  const float* s_proj_w   = (const float*)d_in[30];
  const float* s_proj_b   = (const float*)d_in[31];
  const float* s_ln2_s    = (const float*)d_in[32];
  const float* s_ln2_b    = (const float*)d_in[33];
  const float* s_mlp1_w   = (const float*)d_in[34];
  const float* s_mlp1_b   = (const float*)d_in[35];
  const float* s_mlp2_w   = (const float*)d_in[36];
  const float* s_mlp2_b   = (const float*)d_in[37];
  const float* ln_post_s  = (const float*)d_in[38];
  const float* ln_post_b  = (const float*)d_in[39];
  const float* out_w      = (const float*)d_in[40];
  const float* out_b      = (const float*)d_in[41];

  char* ws = (char*)d_ws;
  size_t off = 0;
  auto alloc = [&](size_t bytes) -> char* {
    char* p = ws + off;
    off += (bytes + 255) & ~(size_t)255;
    return p;
  };

  float* emb  = (float*)alloc((size_t)65536 * 128 * 4);   // 33.5 MB
  h16*   lnkv = (h16*)  alloc((size_t)65536 * 128 * 2);   // 16.8 MB
  h16*   A1   = (h16*)  alloc((size_t)9216 * 1056 * 2);   // 19.5 MB
  h16*   vt   = (h16*)  alloc((size_t)8 * 128 * 8192 * 2);// 16.8 MB
  h16*   y1   = (h16*)  alloc((size_t)9216 * 128 * 2);
  h16*   y2   = (h16*)  alloc((size_t)10240 * 128 * 2);
  float* hbuf = (float*)alloc((size_t)2048 * 128 * 4);
  h16*   lnx  = (h16*)  alloc((size_t)2048 * 128 * 2);
  h16*   qbuf = (h16*)  alloc((size_t)2048 * 128 * 2);
  h16*   attn = (h16*)  alloc((size_t)2048 * 128 * 2);
  h16*   gbuf = (h16*)  alloc((size_t)2048 * 512 * 2);
  h16*   ksb  = (h16*)  alloc((size_t)2048 * 128 * 2);
  h16*   vst  = (h16*)  alloc((size_t)8 * 128 * 256 * 2);
  h16*   lnf  = (h16*)  alloc((size_t)1024 * 128 * 2);
  int*   idxb = (int*)  alloc((size_t)8192 * 4);
  h16*   W1t  = (h16*)  alloc((size_t)128 * 1056 * 2);
  h16*   W2t  = (h16*)  alloc((size_t)128 * 1024 * 2);
  h16*   KVt  = (h16*)  alloc((size_t)256 * 128 * 2);
  h16*   Qt   = (h16*)  alloc((size_t)128 * 128 * 2);
  h16*   Pt   = (h16*)  alloc((size_t)128 * 128 * 2);
  h16*   M1t  = (h16*)  alloc((size_t)512 * 128 * 2);
  h16*   M2t  = (h16*)  alloc((size_t)128 * 512 * 2);
  h16*   QKVt = (h16*)  alloc((size_t)3 * 384 * 128 * 2);
  h16*   Pst  = (h16*)  alloc((size_t)3 * 128 * 128 * 2);
  h16*   M1st = (h16*)  alloc((size_t)3 * 512 * 128 * 2);
  h16*   M2st = (h16*)  alloc((size_t)3 * 128 * 512 * 2);
  h16*   Ot   = (h16*)  alloc((size_t)128 * 128 * 2);
  // Aliases (lifetimes disjoint): A2 reuses emb's region, kbuf reuses A1's.
  h16* A2   = (h16*)emb;   // 21.0 MB <= 33.5 MB, emb dead after im2col1
  h16* kbuf = (h16*)A1;    // 16.8 MB <= 19.5 MB, A1 dead after conv1 GEMM

  const float qscale = 0.17677669529663689f;  // 1/sqrt(32)

  // ---- weight prep
  k_cast_pad<<<(128 * 1056 + 255) / 256, 256, 0, stream>>>(conv1_w, W1t, 1040, 1056, 128);
  k_cast_pad<<<(128 * 1024 + 255) / 256, 256, 0, stream>>>(conv2_w, W2t, 1024, 1024, 128);
  k_tcast<<<(128 * 128 + 255) / 256, 256, 0, stream>>>(c_q_w, Qt, 128, 128);
  k_tcast<<<(128 * 256 + 255) / 256, 256, 0, stream>>>(c_kv_w, KVt, 128, 256);
  k_tcast<<<(128 * 128 + 255) / 256, 256, 0, stream>>>(c_proj_w, Pt, 128, 128);
  k_tcast<<<(128 * 512 + 255) / 256, 256, 0, stream>>>(c_mlp1_w, M1t, 128, 512);
  k_tcast<<<(512 * 128 + 255) / 256, 256, 0, stream>>>(c_mlp2_w, M2t, 512, 128);
  for (int i = 0; i < 3; i++) {
    k_tcast<<<(128 * 384 + 255) / 256, 256, 0, stream>>>(s_qkv_w + (size_t)i * 128 * 384, QKVt + (size_t)i * 384 * 128, 128, 384);
    k_tcast<<<(128 * 128 + 255) / 256, 256, 0, stream>>>(s_proj_w + (size_t)i * 128 * 128, Pst + (size_t)i * 128 * 128, 128, 128);
    k_tcast<<<(128 * 512 + 255) / 256, 256, 0, stream>>>(s_mlp1_w + (size_t)i * 128 * 512, M1st + (size_t)i * 512 * 128, 128, 512);
    k_tcast<<<(512 * 128 + 255) / 256, 256, 0, stream>>>(s_mlp2_w + (size_t)i * 512 * 128, M2st + (size_t)i * 128 * 512, 512, 128);
  }
  k_tcast<<<(128 * 128 + 255) / 256, 256, 0, stream>>>(out_w, Ot, 128, 128);

  // ---- point pipeline
  k_emb<<<4096, 128, 0, stream>>>(points, pe_w, pe_b, c_ln2_s, c_ln2_b, emb, lnkv);
  k_bq<<<256, 128, 0, stream>>>(points, idxb);
  k_im2col1<<<1024, 256, 0, stream>>>(points, emb, idxb, A1);
  gemm(stream, A1, W1t, conv1_b, 9216, 1056, 128, GM_SILU, y1, nullptr, nullptr, nullptr, 0, 0.f);
  k_im2col2<<<1024, 256, 0, stream>>>(y1, A2);
  gemm(stream, A2, W2t, conv2_b, 10240, 1024, 128, GM_SILU, y2, nullptr, nullptr, nullptr, 0, 0.f);
  k_h0<<<512, 128, 0, stream>>>(y2, otok, ln_pre_s, ln_pre_b, hbuf);

  // ---- cross-attention block
  gemm(stream, lnkv, KVt, c_kv_b, 65536, 128, 256, GM_KV, nullptr, nullptr, kbuf, vt, 8192, 0.f);
  k_ln<<<512, 128, 0, stream>>>(hbuf, c_ln1_s, c_ln1_b, lnx, 2048, 0);
  gemm(stream, lnx, Qt, c_q_b, 2048, 128, 128, GM_QS, qbuf, nullptr, nullptr, nullptr, 0, qscale);
  k_attn<<<8 * 4 * 16, 32, 0, stream>>>(qbuf, kbuf, vt, attn, 256, 8192);
  gemm(stream, attn, Pt, c_proj_b, 2048, 128, 128, GM_ADDF, nullptr, hbuf, nullptr, nullptr, 0, 0.f);
  k_ln<<<512, 128, 0, stream>>>(hbuf, c_ln3_s, c_ln3_b, lnx, 2048, 0);
  gemm(stream, lnx, M1t, c_mlp1_b, 2048, 128, 512, GM_GELU, gbuf, nullptr, nullptr, nullptr, 0, 0.f);
  gemm(stream, gbuf, M2t, c_mlp2_b, 2048, 512, 128, GM_ADDF, nullptr, hbuf, nullptr, nullptr, 0, 0.f);

  // ---- self-attention layers
  for (int i = 0; i < 3; i++) {
    k_ln<<<512, 128, 0, stream>>>(hbuf, s_ln1_s + i * 128, s_ln1_b + i * 128, lnx, 2048, 0);
    gemm(stream, lnx, QKVt + (size_t)i * 384 * 128, s_qkv_b + i * 384, 2048, 128, 384,
         GM_QKV, qbuf, nullptr, ksb, vst, 256, qscale);
    k_attn<<<8 * 4 * 16, 32, 0, stream>>>(qbuf, ksb, vst, attn, 256, 256);
    gemm(stream, attn, Pst + (size_t)i * 128 * 128, s_proj_b + i * 128, 2048, 128, 128,
         GM_ADDF, nullptr, hbuf, nullptr, nullptr, 0, 0.f);
    k_ln<<<512, 128, 0, stream>>>(hbuf, s_ln2_s + i * 128, s_ln2_b + i * 128, lnx, 2048, 0);
    gemm(stream, lnx, M1st + (size_t)i * 512 * 128, s_mlp1_b + i * 512, 2048, 128, 512,
         GM_GELU, gbuf, nullptr, nullptr, nullptr, 0, 0.f);
    gemm(stream, gbuf, M2st + (size_t)i * 128 * 512, s_mlp2_b + i * 128, 2048, 512, 128,
         GM_ADDF, nullptr, hbuf, nullptr, nullptr, 0, 0.f);
  }

  // ---- output head: LN over last 128 tokens per batch, projection, tanh
  k_ln<<<256, 128, 0, stream>>>(hbuf, ln_post_s, ln_post_b, lnf, 1024, 1);
  gemm(stream, lnf, Ot, out_b, 1024, 128, 128, GM_TANH, nullptr, (float*)d_out,
       nullptr, nullptr, 0, 0.f);
}
